// TransformerBlock_32066225832176
// MI455X (gfx1250) — compile-verified, matrix operand order checked by reading, not run on hardware
//
#include <hip/hip_runtime.h>
#include <cstdint>

typedef __attribute__((ext_vector_type(16))) _Float16 v16h;
typedef __attribute__((ext_vector_type(8)))  _Float16 v8h;
typedef __attribute__((ext_vector_type(4)))  _Float16 v4h;
typedef __attribute__((ext_vector_type(8)))  float    v8f;

#define EPS_NORM 1e-6f
#define EPS_ATTN 1e-6f

// ---------------------------------------------------------------------------
// Pack a KxN f32 row-major weight matrix into f16 WMMA B-fragment layout:
// tile (tk, tn) covers K rows [tk*32, tk*32+32), cols [tn*16, tn*16+16).
// Lane L holds col n = tn*16 + (L&15); 16 halves h map to
//   k = tk*32 + 8*(L>>4) + (h&7) + ((h>=8)?16:0)
// Stored per-lane contiguous: wp[((tk*TN + tn)*32 + L)*16 + h]
// ---------------------------------------------------------------------------
__global__ void pack_b_kernel(const float* __restrict__ w, _Float16* __restrict__ wp,
                              int K, int N) {
  int idx = blockIdx.x * blockDim.x + threadIdx.x;
  int total = (K / 32) * (N / 16) * 32;
  if (idx >= total) return;
  int lane = idx & 31;
  int tile = idx >> 5;
  int tn = tile % (N / 16);
  int tk = tile / (N / 16);
  int n  = tn * 16 + (lane & 15);
  int kb = tk * 32 + 8 * (lane >> 4);
  _Float16* dst = wp + (size_t)idx * 16;
#pragma unroll
  for (int h = 0; h < 16; ++h) {
    int k = kb + (h & 7) + ((h >= 8) ? 16 : 0);
    dst[h] = (_Float16)w[(size_t)k * N + n];
  }
}

// ---------------------------------------------------------------------------
// Fused rmsnorm1 + QKV GEMM (M=32 tile, N=768, K=256) + elu(q,k)+1.
// 256 threads = 8 waves; wave w: row-frag rf=w>>2 (2x16 rows),
// col group cw=w&3 covering 12 col-frags (12*16=192 cols). 96 WMMAs/wave.
// ---------------------------------------------------------------------------
__global__ __launch_bounds__(256) void qkv_kernel(
    const float* __restrict__ x, const float* __restrict__ wn1,
    const _Float16* __restrict__ wqkvp, _Float16* __restrict__ qkv) {
  __shared__ float red[32][8];
  __shared__ float rstab[32];
  __shared__ _Float16 As[32][264];   // padded rows (16B-aligned strides)

  const int t    = threadIdx.x;
  const int row0 = blockIdx.x * 32;
  const int r    = t >> 3;
  const int c0   = (t & 7) * 32;

  // rmsnorm over each 256-wide row (8 threads/row, 32 elems/thread)
  float xr[32];
  float ss = 0.f;
  const float* xp = x + (size_t)(row0 + r) * 256 + c0;
#pragma unroll
  for (int i = 0; i < 32; ++i) { xr[i] = xp[i]; ss += xr[i] * xr[i]; }
  red[r][t & 7] = ss;
  __syncthreads();
  if (t < 32) {
    float s = 0.f;
#pragma unroll
    for (int j = 0; j < 8; ++j) s += red[t][j];
    rstab[t] = rsqrtf(s * (1.0f / 256.0f) + EPS_NORM);
  }
  __syncthreads();
  {
    float rs = rstab[r];
#pragma unroll
    for (int i = 0; i < 32; ++i)
      As[r][c0 + i] = (_Float16)(xr[i] * rs * wn1[c0 + i]);
  }
  __syncthreads();

  const int w = t >> 5, lane = t & 31;
  const int rf = w >> 2, cw = w & 3;
  v8f acc[12] = {};
  const int m  = rf * 16 + (lane & 15);
  const int kb = 8 * (lane >> 4);
#pragma unroll
  for (int tk = 0; tk < 8; ++tk) {
    v8h lo = *(const v8h*)&As[m][tk * 32 + kb];
    v8h hi = *(const v8h*)&As[m][tk * 32 + kb + 16];
    v16h a = __builtin_shufflevector(lo, hi, 0,1,2,3,4,5,6,7,8,9,10,11,12,13,14,15);
#pragma unroll
    for (int j = 0; j < 12; ++j) {
      v16h b = *(const v16h*)(wqkvp + ((size_t)(tk * 48 + cw * 12 + j) * 32 + lane) * 16);
      acc[j] = __builtin_amdgcn_wmma_f32_16x16x32_f16(
          false, a, false, b, (short)0, acc[j], false, false);
    }
  }
  const int mb = rf * 16 + 8 * (lane >> 4);
#pragma unroll
  for (int j = 0; j < 12; ++j) {
    int n = (cw * 12 + j) * 16 + (lane & 15);
#pragma unroll
    for (int i = 0; i < 8; ++i) {
      float v = acc[j][i];
      if (n < 512) v = (v > 0.f) ? (v + 1.f) : __expf(v);   // elu(v)+1
      qkv[(size_t)(row0 + mb + i) * 768 + n] = (_Float16)v;
    }
  }
}

// ---------------------------------------------------------------------------
// Partial kv / ksum reduction: grid = B*8*CH blocks; block (bh, ch) reduces
// Nseq/CH tokens into a 32x32 f32 partial outer-product sum + 32 ksum.
// ---------------------------------------------------------------------------
__global__ __launch_bounds__(256) void kv_partial_kernel(
    const _Float16* __restrict__ qkv, float* __restrict__ kvpart,
    float* __restrict__ kspart, int Nseq, int CH) {
  __shared__ _Float16 ks[32][40];
  __shared__ _Float16 vs[32][40];
  const int ch = blockIdx.x % CH;
  const int bh = blockIdx.x / CH;
  const int b = bh >> 3, hh = bh & 7;
  const int t = threadIdx.x;
  const int dd = t >> 3;            // d index for compute; token index for load
  const int j4 = (t & 7) * 4;       // m base for compute; col base for load
  const int nloc = Nseq / CH;
  const size_t base = (size_t)b * Nseq * 768 + (size_t)ch * nloc * 768;
  const _Float16* kb = qkv + base + 256 + hh * 32;
  const _Float16* vb = qkv + base + 512 + hh * 32;
  float a0 = 0, a1 = 0, a2 = 0, a3 = 0, kssum = 0;
  for (int c = 0; c < nloc / 32; ++c) {
    size_t ro = (size_t)(c * 32 + dd) * 768;
    *(v4h*)&ks[dd][j4] = *(const v4h*)(kb + ro + j4);
    *(v4h*)&vs[dd][j4] = *(const v4h*)(vb + ro + j4);
    __syncthreads();
#pragma unroll 8
    for (int np = 0; np < 32; ++np) {
      float kd = (float)ks[np][dd];
      a0 += kd * (float)vs[np][j4 + 0];
      a1 += kd * (float)vs[np][j4 + 1];
      a2 += kd * (float)vs[np][j4 + 2];
      a3 += kd * (float)vs[np][j4 + 3];
      kssum += kd;
    }
    __syncthreads();
  }
  float* kp = kvpart + (size_t)blockIdx.x * 1024 + dd * 32 + j4;
  kp[0] = a0; kp[1] = a1; kp[2] = a2; kp[3] = a3;
  if ((t & 7) == 0) kspart[(size_t)blockIdx.x * 32 + dd] = kssum;
}

// ---------------------------------------------------------------------------
// Reduce CH partials per (b,h) and emit kv directly as f16 WMMA B-fragments:
// kv[h] is (d x m); fragment (bh, mh): lane = 16*((d>>3)&1) + (m&15),
// half = (d&7) + (d>=16 ? 8 : 0). Also final f32 ksum.
// ---------------------------------------------------------------------------
__global__ __launch_bounds__(256) void kv_reduce_kernel(
    const float* __restrict__ kvpart, const float* __restrict__ kspart,
    _Float16* __restrict__ kvf, float* __restrict__ ksum, int CH) {
  const int bh = blockIdx.x;
  const int t = threadIdx.x;
#pragma unroll
  for (int j = 0; j < 4; ++j) {
    int e = t * 4 + j;
    int d = e >> 5, mm = e & 31;
    float s = 0.f;
    for (int ch = 0; ch < CH; ++ch)
      s += kvpart[((size_t)bh * CH + ch) * 1024 + e];
    int mh   = mm >> 4;
    int L    = (d >> 3) & 1;
    int lane = 16 * L + (mm & 15);
    int half = (d & 7) + ((d >= 16) ? 8 : 0);
    kvf[((size_t)(bh * 2 + mh) * 32 + lane) * 16 + half] = (_Float16)s;
  }
  if (t < 32) {
    float s = 0.f;
    for (int ch = 0; ch < CH; ++ch)
      s += kspart[((size_t)bh * CH + ch) * 32 + t];
    ksum[(size_t)bh * 32 + t] = s;
  }
}

// ---------------------------------------------------------------------------
// Per 32-token tile: z = 1/(q.ksum+eps); attn = z*(q@kv) via WMMA (A = q
// rows from LDS, B = prepacked kv fragments); then WMMA out-projection
// (32x256 @ 256x256) + b_out + x residual -> h; fused rmsnorm2;
// out = h + w2*h*rsqrt(ms(h)+eps).
// ---------------------------------------------------------------------------
__global__ __launch_bounds__(256) void attn_kernel(
    const _Float16* __restrict__ qkv, const _Float16* __restrict__ kvf,
    const float* __restrict__ ksum, const float* __restrict__ x,
    const _Float16* __restrict__ woutp, const float* __restrict__ b_out,
    const float* __restrict__ wn2, float* __restrict__ out, int Nseq) {
  __shared__ alignas(32) char smem[20096 + 33280];
  float*    kss  = (float*)(smem);                //  256 f32    =  1024
  float*    zz   = (float*)(smem + 1024);         //  32*8 f32   =  1024
  _Float16* attn = (_Float16*)(smem + 2048);      //  32*264 f16 = 16896
  float*    red  = (float*)(smem + 18944);        //  32*8 f32   =  1024
  float*    msr  = (float*)(smem + 19968);        //  32 f32     =   128
  _Float16* qs   = (_Float16*)(smem + 20096);     //  phase 1: 32*264 f16
  float*    hs   = (float*)(smem + 20096);        //  phase 2: 32*260 f32

  const int t = threadIdx.x;
  const int tilesPerB = Nseq / 32;
  const int bb   = blockIdx.x / tilesPerB;
  const int tile = blockIdx.x % tilesPerB;
  const size_t g0 = (size_t)bb * Nseq + (size_t)tile * 32;

  // prefetch the x residual tile (consumed only in the out-proj epilogue)
  __builtin_prefetch(x + g0 * 256 + (size_t)t * 32, 0, 1);

  // load q tile (cols 0..255 of qkv) + ksum
#pragma unroll
  for (int i = 0; i < 32; ++i) {
    int e = t + i * 256, row = e >> 8, col = e & 255;
    qs[row * 264 + col] = qkv[(g0 + row) * 768 + col];
  }
  kss[t] = ksum[(size_t)bb * 256 + t];
  __syncthreads();

  // z for (token = t>>3, head = t&7)
  {
    const int tok = t >> 3, hh = t & 7;
    const _Float16* qrow = qs + tok * 264 + hh * 32;
    const float* ksr = kss + hh * 32;
    float zacc = 0.f;
#pragma unroll
    for (int d = 0; d < 32; ++d) zacc += (float)qrow[d] * ksr[d];
    zz[tok * 8 + hh] = 1.0f / (zacc + EPS_ATTN);
  }
  __syncthreads();

  const int w = t >> 5, lane = t & 31;
  const int rf = w >> 2;
  const int kb = 8 * (lane >> 4);
  const int m  = rf * 16 + (lane & 15);
  const int tok0 = rf * 16 + 8 * (lane >> 4);

  // attn apply via WMMA: wave w handles heads {2*(w&3), 2*(w&3)+1}, both
  // m-halves: 4 WMMAs/wave. Scale by z in the C-fragment epilogue.
  {
    const int hp = (w & 3) * 2;
#pragma unroll
    for (int hi = 0; hi < 2; ++hi) {
      int h = hp + hi;
      v8h lo  = *(const v8h*)&qs[m * 264 + h * 32 + kb];
      v8h hi8 = *(const v8h*)&qs[m * 264 + h * 32 + kb + 16];
      v16h a = __builtin_shufflevector(lo, hi8, 0,1,2,3,4,5,6,7,8,9,10,11,12,13,14,15);
#pragma unroll
      for (int mh = 0; mh < 2; ++mh) {
        v16h b = *(const v16h*)(kvf + ((size_t)((bb * 8 + h) * 2 + mh) * 32 + lane) * 16);
        v8f zero = {};
        v8f acc = __builtin_amdgcn_wmma_f32_16x16x32_f16(
            false, a, false, b, (short)0, zero, false, false);
#pragma unroll
        for (int i = 0; i < 8; ++i) {
          int tokk = tok0 + i;
          attn[tokk * 264 + h * 32 + mh * 16 + (lane & 15)] =
              (_Float16)(acc[i] * zz[tokk * 8 + h]);
        }
      }
    }
  }
  __syncthreads();

  // out-projection GEMM: 32x256 @ 256x256, 8 waves x 4 frags x 8 K-steps
  const int cw = w & 3;
  v8f acc[4] = {};
#pragma unroll
  for (int tk = 0; tk < 8; ++tk) {
    v8h lo = *(const v8h*)&attn[m * 264 + tk * 32 + kb];
    v8h hi = *(const v8h*)&attn[m * 264 + tk * 32 + kb + 16];
    v16h a = __builtin_shufflevector(lo, hi, 0,1,2,3,4,5,6,7,8,9,10,11,12,13,14,15);
#pragma unroll
    for (int j = 0; j < 4; ++j) {
      v16h b = *(const v16h*)(woutp + ((size_t)(tk * 16 + cw * 4 + j) * 32 + lane) * 16);
      acc[j] = __builtin_amdgcn_wmma_f32_16x16x32_f16(
          false, a, false, b, (short)0, acc[j], false, false);
    }
  }
#pragma unroll
  for (int j = 0; j < 4; ++j) {
    int n = (cw * 4 + j) * 16 + (lane & 15);
    float bo = b_out[n];
#pragma unroll
    for (int i = 0; i < 8; ++i) {
      int mm = tok0 + i;
      float hv = acc[j][i] + bo + x[(g0 + mm) * 256 + n];   // residual
      hs[mm * 260 + n] = hv;
    }
  }
  __syncthreads();

  // rmsnorm2 reduction
  {
    const int row = t >> 3, seg = (t & 7) * 32;
    float ss = 0.f;
#pragma unroll
    for (int i = 0; i < 32; ++i) { float v = hs[row * 260 + seg + i]; ss += v * v; }
    red[row * 8 + (t & 7)] = ss;
  }
  __syncthreads();
  if (t < 32) {
    float s = 0.f;
#pragma unroll
    for (int j = 0; j < 8; ++j) s += red[t * 8 + j];
    msr[t] = rsqrtf(s * (1.0f / 256.0f) + EPS_NORM);
  }
  __syncthreads();
#pragma unroll
  for (int i = 0; i < 32; ++i) {
    int e = t + i * 256, row = e >> 8, col = e & 255;
    float hv = hs[row * 260 + col];
    out[(g0 + row) * 256 + col] = hv + wn2[col] * hv * msr[row];
  }
}

// ---------------------------------------------------------------------------
extern "C" void kernel_launch(void* const* d_in, const int* in_sizes, int n_in,
                              void* d_out, int out_size, void* d_ws, size_t ws_size,
                              hipStream_t stream) {
  const float* x    = (const float*)d_in[0];
  const float* wn1  = (const float*)d_in[1];
  const float* wqkv = (const float*)d_in[2];
  const float* wout = (const float*)d_in[3];
  const float* bout = (const float*)d_in[4];
  const float* wn2  = (const float*)d_in[5];
  float* out = (float*)d_out;

  const int D = 256;
  const int TOK = in_sizes[0] / D;      // B*N = 131072
  const int Nseq = 8192;
  const int B = TOK / Nseq;
  const int CH = 8;                     // kv reduction chunks per (b,h)

  char* ws = (char*)d_ws;
  _Float16* wqkvp  = (_Float16*)(ws);                         // 384 KB
  _Float16* woutp  = (_Float16*)(ws + (size_t)(384 << 10));   // 128 KB
  float*    ksbuf  = (float*)(ws + (size_t)(512 << 10));      //  16 KB
  _Float16* kvf    = (_Float16*)(ws + (size_t)(576 << 10));   // 256 KB
  float*    kspart = (float*)(ws + (size_t)(832 << 10));      // 128 KB
  float*    kvpart = (float*)(ws + (size_t)(1024 << 10));     //   4 MB
  _Float16* qkv    = (_Float16*)(ws + (size_t)(8192 << 10));  // 192 MB (f16)

  pack_b_kernel<<<(8 * 48 * 32 + 255) / 256, 256, 0, stream>>>(wqkv, wqkvp, 256, 768);
  pack_b_kernel<<<(8 * 16 * 32 + 255) / 256, 256, 0, stream>>>(wout, woutp, 256, 256);
  qkv_kernel<<<TOK / 32, 256, 0, stream>>>(x, wn1, wqkvp, qkv);
  kv_partial_kernel<<<B * 8 * CH, 256, 0, stream>>>(qkv, kvpart, kspart, Nseq, CH);
  kv_reduce_kernel<<<B * 8, 256, 0, stream>>>(kvpart, kspart, kvf, ksbuf, CH);
  attn_kernel<<<TOK / 32, 256, 0, stream>>>(qkv, kvf, ksbuf, x, woutp, bout, wn2, out, Nseq);
}
